// CTResNetNeck_1254130450971
// MI455X (gfx1250) — matrix operand order repaired and checked by reading, not compiled, not run on hardware
//
#include <hip/hip_runtime.h>
#include <hip/hip_bf16.h>
#include <math.h>

typedef __attribute__((ext_vector_type(16))) _Float16 v16h;
typedef __attribute__((ext_vector_type(8)))  _Float16 v8h;
typedef __attribute__((ext_vector_type(8)))  float    v8f;

#define BN_EPS 1e-5f
#define MBLK 4   // 4 x 16 = 64 output rows per wave

// ---------------------------------------------------------------------------
// f32 -> f16 pack (used for MDCN weights, which are already [OC][C*9] flat)
// ---------------------------------------------------------------------------
__global__ void k_f32_to_f16(const float* __restrict__ s, _Float16* __restrict__ d, int n) {
  int i = blockIdx.x * blockDim.x + threadIdx.x;
  if (i < n) d[i] = (_Float16)s[i];
}

// ---------------------------------------------------------------------------
// 3x3 pad-1 conv producing the 27-channel offset/mask predictor (fp32)
// out layout: [B][27][H*W]
// ---------------------------------------------------------------------------
__global__ void k_predconv(const float* __restrict__ x, const float* __restrict__ w,
                           const float* __restrict__ bias, float* __restrict__ out,
                           int B, int C, int H, int W) {
  int HW = H * W;
  int tid = blockIdx.x * blockDim.x + threadIdx.x;
  int total = B * 27 * HW;
  if (tid >= total) return;
  int p = tid % HW;
  int o = (tid / HW) % 27;
  int b = tid / (27 * HW);
  int y = p / W, xx = p % W;
  const float* xb = x + (size_t)b * C * HW;
  const float* wo = w + (size_t)o * C * 9;
  float acc = bias[o];
  for (int c = 0; c < C; ++c) {
    const float* xc = xb + (size_t)c * HW;
    const float* wc = wo + (size_t)c * 9;
#pragma unroll
    for (int ky = 0; ky < 3; ++ky) {
      int iy = y + ky - 1;
      if (iy < 0 || iy >= H) continue;
#pragma unroll
      for (int kx = 0; kx < 3; ++kx) {
        int ix = xx + kx - 1;
        if (ix < 0 || ix >= W) continue;
        acc += wc[ky * 3 + kx] * xc[iy * W + ix];
      }
    }
  }
  out[tid] = acc;
}

// ---------------------------------------------------------------------------
// Modulated-deformable sampling -> f16 cols matrix.
// cols layout: [B][H*W][C*9]  (pixel-major, K contiguous => WMMA B operand)
// pred layout: [B][27][H*W]; channels 2k = dy, 2k+1 = dx, 18+k = mask logit.
// ---------------------------------------------------------------------------
__global__ void k_mdcn_cols(const float* __restrict__ x, const float* __restrict__ pred,
                            _Float16* __restrict__ cols, int B, int C, int H, int W) {
  int HW = H * W;
  int tid = blockIdx.x * blockDim.x + threadIdx.x;
  int total = B * HW * 9;
  if (tid >= total) return;
  int t = tid % 9;
  int p = (tid / 9) % HW;
  int b = tid / (9 * HW);
  int y = p / W, xq = p % W;

  const float* pb = pred + (size_t)b * 27 * HW;
  float dy = pb[(size_t)(2 * t) * HW + p];
  float dx = pb[(size_t)(2 * t + 1) * HW + p];
  float mk = pb[(size_t)(18 + t) * HW + p];
  mk = 1.0f / (1.0f + __expf(-mk));

  float py = dy + (float)y + (float)(t / 3) - 1.0f;
  float px = dx + (float)xq + (float)(t % 3) - 1.0f;
  float y0f = floorf(py), x0f = floorf(px);
  float wy = py - y0f, wx = px - x0f;
  int y0 = (int)y0f, x0 = (int)x0f;
  int y1 = y0 + 1, x1 = x0 + 1;

  float vy0 = (y0 >= 0 && y0 < H) ? 1.0f : 0.0f;
  float vy1 = (y1 >= 0 && y1 < H) ? 1.0f : 0.0f;
  float vx0 = (x0 >= 0 && x0 < W) ? 1.0f : 0.0f;
  float vx1 = (x1 >= 0 && x1 < W) ? 1.0f : 0.0f;
  float w00 = (1.0f - wy) * (1.0f - wx) * vy0 * vx0 * mk;
  float w01 = (1.0f - wy) * wx * vy0 * vx1 * mk;
  float w10 = wy * (1.0f - wx) * vy1 * vx0 * mk;
  float w11 = wy * wx * vy1 * vx1 * mk;

  int yc0 = y0 < 0 ? 0 : (y0 > H - 1 ? H - 1 : y0);
  int yc1 = y1 < 0 ? 0 : (y1 > H - 1 ? H - 1 : y1);
  int xc0 = x0 < 0 ? 0 : (x0 > W - 1 ? W - 1 : x0);
  int xc1 = x1 < 0 ? 0 : (x1 > W - 1 ? W - 1 : x1);
  int i00 = yc0 * W + xc0, i01 = yc0 * W + xc1;
  int i10 = yc1 * W + xc0, i11 = yc1 * W + xc1;

  const float* xb = x + (size_t)b * C * HW;
  _Float16* cp = cols + ((size_t)b * HW + p) * (size_t)(C * 9) + t;
  for (int c = 0; c < C; ++c) {
    const float* xc = xb + (size_t)c * HW;
    float v = xc[i00] * w00 + xc[i01] * w01 + xc[i10] * w10 + xc[i11] * w11;
    cp[(size_t)c * 9] = (_Float16)v;
  }
}

// ---------------------------------------------------------------------------
// Deconv (k=4,s=2,p=1) cols per parity class.
// cols layout: [q][B][Hin*Win][C*4], k = c*4 + dy*2 + dx, tap: iy=y+qy-dy.
// ---------------------------------------------------------------------------
__global__ void k_deconv_cols(const float* __restrict__ in, _Float16* __restrict__ cols,
                              int B, int C, int Hin, int Win) {
  int HW = Hin * Win;
  int tid = blockIdx.x * blockDim.x + threadIdx.x;
  int total = B * 4 * HW;
  if (tid >= total) return;
  int p = tid % HW;
  int b = (tid / HW) % B;
  int q = tid / (HW * B);
  int qy = q >> 1, qx = q & 1;
  int y = p / Win, x = p % Win;
  int iby = y + qy, ibx = x + qx;

  const float* ib = in + (size_t)b * C * HW;
  _Float16* cp = cols + ((size_t)(q * B + b) * HW + p) * (size_t)(C * 4);
  for (int c = 0; c < C; ++c) {
    const float* icp = ib + (size_t)c * HW;
#pragma unroll
    for (int dy = 0; dy < 2; ++dy) {
#pragma unroll
      for (int dx = 0; dx < 2; ++dx) {
        int iy = iby - dy, ix = ibx - dx;
        bool v = (iy >= 0 && iy < Hin && ix >= 0 && ix < Win);
        cp[(size_t)c * 4 + dy * 2 + dx] = (_Float16)(v ? icp[iy * Win + ix] : 0.0f);
      }
    }
  }
}

// ---------------------------------------------------------------------------
// Deconv weight repack per parity class:
// wa[q][co][ci*4 + dy*2+dx] = tw[ci][co][(1-qy)+2dy][(1-qx)+2dx]   (f16)
// tw layout: [C][C][4][4]  (torch ConvTranspose2d: [Cin][Cout][kH][kW])
// ---------------------------------------------------------------------------
__global__ void k_deconv_pack_w(const float* __restrict__ tw, _Float16* __restrict__ wa, int C) {
  int tid = blockIdx.x * blockDim.x + threadIdx.x;
  int total = 4 * C * C * 4;
  if (tid >= total) return;
  int t = tid & 3;
  int dy = t >> 1, dx = t & 1;
  int ci = (tid >> 2) % C;
  int co = ((tid >> 2) / C) % C;
  int q  = (tid >> 2) / (C * C);
  int qy = q >> 1, qx = q & 1;
  int ky = (1 - qy) + 2 * dy;
  int kx = (1 - qx) + 2 * dx;
  wa[((size_t)(q * C + co)) * (size_t)(C * 4) + (size_t)ci * 4 + t] =
      (_Float16)tw[(((size_t)ci * C + co) * 4 + ky) * 4 + kx];
}

// ---------------------------------------------------------------------------
// WMMA f16 GEMM, register-blocked along M:
//   one wave computes a 64x16 tile (MBLK=4 sub-tiles); per K-step it loads
//   ONE B fragment and MBLK A fragments -> MBLK wmma ops (4x less B traffic).
// D[m][n] = sum_k A[m][k] * Bc[n][k], fused BN(+bias)+ReLU epilogue.
// A row-major [M][K]; Bc pixel-major [N][K] (K contiguous).
// Fragment layouts follow the CDNA5 ISA 16-bit A (16x32) / B (32x16) tables.
// mode 0: out[(b*OC+m)*HW + p],            n = b*HW + p
// mode 1: out[((b*OC+m)*Hout + 2y+qy)*Wout + 2x+qx],  n = b*Hs*Ws + y*Ws + x
// ---------------------------------------------------------------------------
__global__ __launch_bounds__(32)
void k_wmma_gemm(const _Float16* __restrict__ A, const _Float16* __restrict__ Bc,
                 float* __restrict__ out, int K,
                 const float* __restrict__ gS, const float* __restrict__ bS,
                 const float* __restrict__ mS, const float* __restrict__ vS,
                 const float* __restrict__ bias,
                 int mode, int OC, int HW, int Hs, int Ws, int Hout, int Wout,
                 int qy, int qx) {
  int lane = threadIdx.x & 31;
  int row = lane & 15;
  int hi = lane >> 4;            // half-wave select per ISA fragment tables
  int mt = blockIdx.x, nt = blockIdx.y;

  const size_t subStride = (size_t)16 * K;   // rows between M sub-tiles
  const _Float16* arow = A + ((size_t)(mt * (16 * MBLK) + row)) * K + hi * 8;
  const _Float16* brow = Bc + ((size_t)(nt * 16 + row)) * K + hi * 8;

  v8f acc[MBLK];
#pragma unroll
  for (int s = 0; s < MBLK; ++s) acc[s] = (v8f){};

  for (int k = 0; k < K; k += 32) {
    __builtin_prefetch((const void*)(brow + k + 128), 0, 0);
    // B 32x16: lanes<16 hold K 0-7 & 16-23 of column n; lanes>=16: 8-15 & 24-31
    v8h b0 = *(const v8h*)(brow + k);
    v8h b1 = *(const v8h*)(brow + k + 16);
    v16h bv = __builtin_shufflevector(b0, b1, 0, 1, 2, 3, 4, 5, 6, 7,
                                      8, 9, 10, 11, 12, 13, 14, 15);
#pragma unroll
    for (int s = 0; s < MBLK; ++s) {
      // A 16x32: lanes<16 hold K 0-7 & 16-23 of their row; lanes>=16: 8-15 & 24-31
      const _Float16* ap = arow + (size_t)s * subStride + k;
      v8h a0 = *(const v8h*)(ap);
      v8h a1 = *(const v8h*)(ap + 16);
      v16h av = __builtin_shufflevector(a0, a1, 0, 1, 2, 3, 4, 5, 6, 7,
                                        8, 9, 10, 11, 12, 13, 14, 15);
      acc[s] = __builtin_amdgcn_wmma_f32_16x16x32_f16(false, av, false, bv,
                                                      (short)0, acc[s], false, false);
    }
  }

  int n = nt * 16 + row;
  int b_, p_, y_, x_;
  if (mode == 0) {
    b_ = n / HW; p_ = n % HW;
  } else {
    int hw = Hs * Ws;
    b_ = n / hw; p_ = n % hw;
    y_ = p_ / Ws; x_ = p_ % Ws;
  }
#pragma unroll
  for (int s = 0; s < MBLK; ++s) {
#pragma unroll
    for (int v = 0; v < 8; ++v) {
      int m = mt * (16 * MBLK) + s * 16 + v + hi * 8;  // C/D: lanes>=16 -> M+8
      float sc = gS[m] * rsqrtf(vS[m] + BN_EPS);
      float tt = bS[m] - mS[m] * sc;
      if (bias) tt += bias[m] * sc;
      float r = fmaxf(acc[s][v] * sc + tt, 0.0f);
      size_t idx;
      if (mode == 0) {
        idx = ((size_t)b_ * OC + m) * HW + p_;
      } else {
        int oy = 2 * y_ + qy, ox = 2 * x_ + qx;
        idx = (((size_t)b_ * OC + m) * Hout + oy) * Wout + ox;
      }
      out[idx] = r;
    }
  }
}

// ---------------------------------------------------------------------------
// Launcher
// ---------------------------------------------------------------------------
extern "C" void kernel_launch(void* const* d_in, const int* in_sizes, int n_in,
                              void* d_out, int out_size, void* d_ws, size_t ws_size,
                              hipStream_t stream) {
  (void)in_sizes; (void)n_in; (void)out_size; (void)ws_size;
  const int B = 16;

  const float* x = (const float*)d_in[0];
  const float* ow[3] = {(const float*)d_in[1], (const float*)d_in[6],  (const float*)d_in[11]};
  const float* ob[3] = {(const float*)d_in[2], (const float*)d_in[7],  (const float*)d_in[12]};
  const float* dw[3] = {(const float*)d_in[3], (const float*)d_in[8],  (const float*)d_in[13]};
  const float* tw[3] = {(const float*)d_in[4], (const float*)d_in[9],  (const float*)d_in[14]};
  const float* tb[3] = {(const float*)d_in[5], (const float*)d_in[10], (const float*)d_in[15]};
  const float *G[6], *Bb[6], *Mm[6], *Vv[6];
  for (int j = 0; j < 6; ++j) {
    G[j]  = (const float*)d_in[16 + 4 * j + 0];
    Bb[j] = (const float*)d_in[16 + 4 * j + 1];
    Mm[j] = (const float*)d_in[16 + 4 * j + 2];
    Vv[j] = (const float*)d_in[16 + 4 * j + 3];
  }

  // workspace carve-out
  char* ws = (char*)d_ws;
  size_t off = 0;
  auto carve = [&](size_t bytes) -> void* {
    void* p = ws + off;
    off += (bytes + 255) & ~(size_t)255;
    return p;
  };
  float*    pred = (float*)   carve((size_t)B * 27 * 4096 * 4);        // max stage-3
  float*    a1   = (float*)   carve((size_t)B * 256 * 256 * 4);        // [16,256,16,16]
  float*    u1   = (float*)   carve((size_t)B * 256 * 1024 * 4);       // [16,256,32,32]
  float*    a2   = (float*)   carve((size_t)B * 128 * 1024 * 4);       // [16,128,32,32]
  float*    u2   = (float*)   carve((size_t)B * 128 * 4096 * 4);       // [16,128,64,64]
  float*    a3   = (float*)   carve((size_t)B * 64 * 4096 * 4);        // [16,64,64,64]
  _Float16* cols = (_Float16*)carve((size_t)B * 4096 * 1152 * 2);      // max cols matrix
  _Float16* wbuf = (_Float16*)carve((size_t)256 * 4608 * 2);           // max weight pack

  const int TB = 256;
  auto ceil_div = [](int a, int b) { return (a + b - 1) / b; };

  // stage descriptors: {Cin, H, W, OC}
  struct St { int C, H, W, OC; };
  const St st[3] = {{512, 16, 16, 256}, {256, 32, 32, 128}, {128, 64, 64, 64}};
  const float* stage_in[3] = {x, u1, u2};
  float* stage_out[3] = {a1, a2, a3};
  float* up_out[3] = {u1, u2, (float*)d_out};

  for (int i = 0; i < 3; ++i) {
    const int C = st[i].C, H = st[i].H, W = st[i].W, OC = st[i].OC;
    const int HW = H * W, K9 = C * 9;
    const float* in = stage_in[i];

    // --- MDCN stage (BN index 2*i) ---
    k_predconv<<<ceil_div(B * 27 * HW, TB), TB, 0, stream>>>(in, ow[i], ob[i], pred, B, C, H, W);
    k_mdcn_cols<<<ceil_div(B * HW * 9, TB), TB, 0, stream>>>(in, pred, cols, B, C, H, W);
    k_f32_to_f16<<<ceil_div(OC * K9, TB), TB, 0, stream>>>(dw[i], wbuf, OC * K9);
    {
      dim3 grid(OC / (16 * MBLK), (B * HW) / 16);
      k_wmma_gemm<<<grid, 32, 0, stream>>>(wbuf, cols, stage_out[i], K9,
                                           G[2 * i], Bb[2 * i], Mm[2 * i], Vv[2 * i],
                                           nullptr, 0, OC, HW, 0, 0, 0, 0, 0, 0);
    }

    // --- Deconv up-stage (BN index 2*i+1); Cin = Cout = OC, Hin = H ---
    const int DC = OC, Hin = H, Win = W, DHW = Hin * Win, DK = DC * 4;
    k_deconv_cols<<<ceil_div(B * 4 * DHW, TB), TB, 0, stream>>>(stage_out[i], cols, B, DC, Hin, Win);
    k_deconv_pack_w<<<ceil_div(4 * DC * DC * 4, TB), TB, 0, stream>>>(tw[i], wbuf, DC);
    for (int q = 0; q < 4; ++q) {
      dim3 grid(DC / (16 * MBLK), (B * DHW) / 16);
      const _Float16* Aq = wbuf + (size_t)q * DC * DK;
      const _Float16* Bq = cols + (size_t)q * B * DHW * DK;
      k_wmma_gemm<<<grid, 32, 0, stream>>>(Aq, Bq, up_out[i], DK,
                                           G[2 * i + 1], Bb[2 * i + 1], Mm[2 * i + 1], Vv[2 * i + 1],
                                           tb[i], 1, DC, DHW, Hin, Win,
                                           2 * Hin, 2 * Win, q >> 1, q & 1);
    }
  }
}
